// AttentionAggregator_4140348473475
// MI455X (gfx1250) — matrix operand order repaired and checked by reading, not compiled, not run on hardware
//
#include <hip/hip_runtime.h>
#include <hip/hip_bf16.h>
#include <stdint.h>

typedef __bf16 v16bf __attribute__((ext_vector_type(16)));
typedef float  v8f   __attribute__((ext_vector_type(8)));

#define B_ROWS 4096
#define NGENES 4096
#define NGROUP 1024
#define GSZ    32
#define TTOT   (NGROUP*GSZ)

__device__ __forceinline__ unsigned short f32_to_bf16_rne(float x) {
  union { float f; unsigned u; } c; c.f = x;
  unsigned u = c.u;
  u += 0x7FFFu + ((u >> 16) & 1u);   // round-to-nearest-even
  return (unsigned short)(u >> 16);
}

// ---------------------------------------------------------------------------
// Kernel 1: per-group (32-wide) softmax of att_weights -> bf16 attn in ws.
// One wave32 per group; cross-lane reduce via shuffles.
// ---------------------------------------------------------------------------
__global__ __launch_bounds__(256) void attn_softmax_kernel(
    const float* __restrict__ att, unsigned short* __restrict__ attn_bf16) {
  const int lane = threadIdx.x & 31;
  const int wave = threadIdx.x >> 5;
  const int g = blockIdx.x * 8 + wave;
  float v = att[g * GSZ + lane];
  float m = v;
  #pragma unroll
  for (int off = 16; off > 0; off >>= 1)
    m = fmaxf(m, __shfl_xor(m, off, 32));
  float ex = __expf(v - m);
  float s = ex;
  #pragma unroll
  for (int off = 16; off > 0; off >>= 1)
    s += __shfl_xor(s, off, 32);
  attn_bf16[g * GSZ + lane] = f32_to_bf16_rne(ex / s);
}

// ---------------------------------------------------------------------------
// Kernel 2: grid = 128 pair-tiles (32 b-rows) x 2 group-halves (512 groups).
// LDS feature slab packs two b-rows per 32-bit word: {F[b,c], F[b+16,c]} as
// bf16; one ds_load_b32 gather feeds TWO A-fragments, split out with
// v_perm_b32 (one VALU op per element). Two chained WMMAs share the group's
// indices and B-fragment select.
// ---------------------------------------------------------------------------
__global__ __launch_bounds__(512, 1) void attn_aggregate_wmma_kernel(
    const float* __restrict__ feat, const int* __restrict__ flat_idx,
    const unsigned short* __restrict__ attn_bf16, float* __restrict__ out) {
  extern __shared__ char smemRaw[];
  unsigned* sfeat = (unsigned*)smemRaw;                       // 16*4096 words = 256 KB
  int* sIdxAll = (int*)(smemRaw + 16 * NGENES * sizeof(unsigned)); // 16 waves * 512 ints

  const int tid  = threadIdx.x;
  const int lane = tid & 31;
  const int wave = tid >> 5;
  const int pair = blockIdx.x >> 1;   // 128 slabs of 32 b-rows
  const int half = blockIdx.x & 1;    // which 512 groups

  // ---- Stage 32-row f32 slab as packed bf16-pair words in LDS ----
  // word[m*NGENES + c] = { bf16 F[pair*32+m, c], bf16 F[pair*32+16+m, c] }
  {
    const float* fbase = feat + (size_t)pair * 32 * NGENES;
    for (int j = tid; j < 16 * (NGENES / 4); j += 512) {
      const int m  = j >> 10;           // row 0..15
      const int jc = j & 1023;          // float4 index within row
      float4 va = *(const float4*)(fbase + (size_t)m * NGENES + jc * 4);
      float4 vb = *(const float4*)(fbase + (size_t)(m + 16) * NGENES + jc * 4);
      uint4 w;
      w.x = (unsigned)f32_to_bf16_rne(va.x) | ((unsigned)f32_to_bf16_rne(vb.x) << 16);
      w.y = (unsigned)f32_to_bf16_rne(va.y) | ((unsigned)f32_to_bf16_rne(vb.y) << 16);
      w.z = (unsigned)f32_to_bf16_rne(va.z) | ((unsigned)f32_to_bf16_rne(vb.z) << 16);
      w.w = (unsigned)f32_to_bf16_rne(va.w) | ((unsigned)f32_to_bf16_rne(vb.w) << 16);
      *(uint4*)(sfeat + (size_t)m * NGENES + jc * 4) = w;
    }
  }
  __syncthreads();

  const __bf16* attnb = (const __bf16*)attn_bf16;
  int* sIdx = sIdxAll + wave * 512;
  const unsigned sIdxByte = (unsigned)(16 * NGENES * 4 + wave * 512 * 4);

  const int n    = lane & 15;       // A row m, B/D column
  const int hi   = lane >> 4;       // lane half
  const int koff = hi * 16;         // A-frag K offset for this half
  const int mb   = n * NGENES;      // word base for row m in LDS

  v16bf bz = {};

  // 32 batches of 16 groups in this half; 16 waves -> 2 batches each.
  for (int bl = wave; bl < 32; bl += 16) {
    const int gbase = (half * 32 + bl) * 16;

    // ---- Async-stage this batch's 512 gather indices into LDS ----
    {
      const int* gp = flat_idx + gbase * GSZ + lane * 16;
      const unsigned lo = sIdxByte + (unsigned)(lane * 16) * 4u;
      #pragma unroll
      for (int q = 0; q < 4; ++q) {
        unsigned long long ga = (unsigned long long)(uintptr_t)(gp + q * 4);
        unsigned l = lo + (unsigned)q * 16u;
        asm volatile("global_load_async_to_lds_b128 %0, %1, off"
                     :: "v"(l), "v"(ga) : "memory");
      }
    }

    // ---- Preload this lane's B-fragment attn values (group gbase+n) ----
    // B layout: lane L -> column L%16, element e -> K = e + hi*16.
    const int gsel = gbase + n;
    v16bf bat = *(const v16bf*)(attnb + gsel * GSZ + hi * 16);

    asm volatile("s_wait_asynccnt 0" ::: "memory");

    v8f acc0 = {};   // rows pair*32 + 0..15
    v8f acc1 = {};   // rows pair*32 + 16..31
    #pragma unroll 4
    for (int i = 0; i < 16; ++i) {
      // A layout: lane L -> row L%16, element e -> K = e + hi*16.
      const int* sI = sIdx + i * 32 + koff;
      int4 i0 = *(const int4*)(sI);
      int4 i1 = *(const int4*)(sI + 4);
      int4 i2 = *(const int4*)(sI + 8);
      int4 i3 = *(const int4*)(sI + 12);

      union { unsigned u[8]; v16bf v; } A0, A1;
      // Two pair-words -> one dword of each fragment via byte-permute:
      //   A0 dword p = {w0.lo16, w1.lo16}   (tile-0 rows)
      //   A1 dword p = {w0.hi16, w1.hi16}   (tile-1 rows)
      #define GPAIR(p, ia, ib)                                        \
        { unsigned w0 = sfeat[mb + (ia)];                             \
          unsigned w1 = sfeat[mb + (ib)];                             \
          A0.u[p] = __builtin_amdgcn_perm(w1, w0, 0x05040100u);       \
          A1.u[p] = __builtin_amdgcn_perm(w1, w0, 0x07060302u); }
      GPAIR(0, i0.x, i0.y) GPAIR(1, i0.z, i0.w)
      GPAIR(2, i1.x, i1.y) GPAIR(3, i1.z, i1.w)
      GPAIR(4, i2.x, i2.y) GPAIR(5, i2.z, i2.w)
      GPAIR(6, i3.x, i3.y) GPAIR(7, i3.z, i3.w)
      #undef GPAIR

      v16bf bfrag = (n == i) ? bat : bz;   // non-zero only in column i
      acc0 = __builtin_amdgcn_wmma_f32_16x16x32_bf16(
          false, A0.v, false, bfrag, (short)0, acc0, false, false);
      acc1 = __builtin_amdgcn_wmma_f32_16x16x32_bf16(
          false, A1.v, false, bfrag, (short)0, acc1, false, false);
    }

    // ---- Store both D tiles: VGPR r -> M = r + hi*8, N = lane%16 ----
    float* orow0 = out + (size_t)(pair * 32 + hi * 8) * NGROUP + gbase + n;
    float* orow1 = out + (size_t)(pair * 32 + 16 + hi * 8) * NGROUP + gbase + n;
    #pragma unroll
    for (int r = 0; r < 8; ++r) {
      orow0[(size_t)r * NGROUP] = acc0[r];
      orow1[(size_t)r * NGROUP] = acc1[r];
    }
  }
}

extern "C" void kernel_launch(void* const* d_in, const int* in_sizes, int n_in,
                              void* d_out, int out_size, void* d_ws, size_t ws_size,
                              hipStream_t stream) {
  const float* feat = (const float*)d_in[0];     // (B, NG) f32
  const float* att  = (const float*)d_in[1];     // (T,) f32
  const int*   fidx = (const int*)d_in[2];       // (T,) int
  unsigned short* attn_bf16 = (unsigned short*)d_ws;  // T bf16 = 64 KB scratch
  float* out = (float*)d_out;                    // (B, G) f32

  attn_softmax_kernel<<<NGROUP / 8, 256, 0, stream>>>(att, attn_bf16);

  const size_t ldsBytes = 16 * NGENES * sizeof(unsigned) + 16 * 512 * sizeof(int);
  attn_aggregate_wmma_kernel<<<(B_ROWS / 32) * 2, 512, ldsBytes, stream>>>(
      feat, fidx, attn_bf16, out);
}
